// DiffPQ_11665131176038
// MI455X (gfx1250) — compile-verified, matrix operand order checked by reading, not run on hardware
//
#include <hip/hip_runtime.h>

typedef __attribute__((ext_vector_type(16))) _Float16 v16h;
typedef __attribute__((ext_vector_type(4)))  _Float16 h4;
typedef __attribute__((ext_vector_type(8)))  float    v8f;

#define B_TOT   16384
#define M_SUB   16
#define K_CENT  256
#define D_SUB   32
#define D_TOT   512
#define ROWS_PB 128
#define THREADS 256
#define NBLK    ((B_TOT / ROWS_PB) * M_SUB)   // 2048
#define PADH    40   // padded row stride in halves (80 B): 16B-aligned, conflict-free b128

// output slot offsets (floats), tuple order: X_r, X_p, X_r_m, X, center, label, output
#define OFF_XR   0u
#define OFF_XP   8388608u
#define OFF_XRM  16777216u
#define OFF_X    25165824u
#define OFF_CTR  33554432u
#define OFF_LBL  33685504u
#define OFF_OUT  33947648u

// ---------------------------------------------------------------------------
// Kernel 1: convert center -> f16 (ws), compute c_sq (ws), copy center -> out
// one wave per (m,k) row of 32 elements
__global__ __launch_bounds__(256) void prep_center(const float* __restrict__ center,
                                                   _Float16* __restrict__ ch,
                                                   float* __restrict__ csq,
                                                   float* __restrict__ out) {
    const int tid  = threadIdx.x;
    const int mk   = blockIdx.x * 8 + (tid >> 5);   // 0..4095
    const int lane = tid & 31;
    const float c  = center[mk * D_SUB + lane];
    out[OFF_CTR + mk * D_SUB + lane] = c;
    ch[mk * D_SUB + lane] = (_Float16)c;
    float s = c * c;
    #pragma unroll
    for (int off = 16; off >= 1; off >>= 1) s += __shfl_xor(s, off, 32);
    if (lane == 0) csq[mk] = s;
}

// ---------------------------------------------------------------------------
// Kernel 2: X -> X_p slot and X slot (pure bandwidth, float4)
__global__ __launch_bounds__(256) void copy_x(const float4* __restrict__ X4,
                                              float* __restrict__ out) {
    const unsigned i = blockIdx.x * 256u + threadIdx.x;   // exactly 2097152 lanes
    const float4 v = X4[i];
    ((float4*)(out + OFF_XP))[i] = v;
    ((float4*)(out + OFF_X))[i]  = v;
}

// ---------------------------------------------------------------------------
// Kernel 3: fused dot(WMMA) -> argmin -> gather -> X_r / X_r_m / label / err
__global__ __launch_bounds__(THREADS) void diffpq_compute(const float* __restrict__ X,
                                                          const float* __restrict__ center,
                                                          const _Float16* __restrict__ center_h,
                                                          const float* __restrict__ csq_g,
                                                          float* __restrict__ out,
                                                          float* __restrict__ partials) {
    __shared__ _Float16 sC[K_CENT * PADH];      // 20 KB: center[m] f16, padded rows
    __shared__ _Float16 sX[ROWS_PB * PADH];     // 10 KB: X tile f16, padded rows
    __shared__ float    sCsq[K_CENT];
    __shared__ int      sLab[ROWS_PB];
    __shared__ float    sErr[8];

    const int tid   = threadIdx.x;
    const int m     = blockIdx.x & (M_SUB - 1);
    const int brow0 = (blockIdx.x >> 4) * ROWS_PB;

    // stage center[m] (already f16 in ws): 4 x 16B chunks per 64B row -> 80B padded rows
    {
        const uint4* src = (const uint4*)(center_h + (size_t)m * K_CENT * D_SUB);
        uint4* dst = (uint4*)sC;
        #pragma unroll
        for (int i = tid; i < K_CENT * 4; i += THREADS)
            dst[(i >> 2) * 5 + (i & 3)] = src[i];
    }
    if (tid < K_CENT) sCsq[tid] = csq_g[m * K_CENT + tid];
    // stage X tile: float4 coalesced loads, fp32 -> f16, packed 8B LDS stores
    {
        const float4* X4 = (const float4*)(X + (size_t)brow0 * D_TOT + m * D_SUB);
        #pragma unroll
        for (int i = tid; i < ROWS_PB * 8; i += THREADS) {
            const int r = i >> 3, c4 = i & 7;
            const float4 v = X4[(size_t)r * (D_TOT / 4) + c4];
            h4 h = { (_Float16)v.x, (_Float16)v.y, (_Float16)v.z, (_Float16)v.w };
            *(h4*)&sX[r * PADH + c4 * 4] = h;
        }
    }
    __syncthreads();

    const int w    = tid >> 5;
    const int lane = tid & 31;
    const int col  = lane & 15;
    const int kb   = (lane < 16) ? 0 : 8;   // 16-bit operand layout: K base per lane half

    // A matrix: 16 rows of this wave's b-tile, full K=32 (=d)
    v16h a;
    {
        const _Float16* p = &sX[(w * 16 + col) * PADH + kb];
        #pragma unroll
        for (int i = 0; i < 8; ++i) { a[i] = p[i]; a[i + 8] = p[i + 16]; }
    }

    float minv[8];
    int   mink[8];
    #pragma unroll
    for (int j = 0; j < 8; ++j) { minv[j] = 3.4e38f; mink[j] = 0; }

    // 16 N-tiles of 16 centroids each; consume accumulators immediately
    #pragma unroll
    for (int t = 0; t < 16; ++t) {
        v16h b;
        const _Float16* p = &sC[(t * 16 + col) * PADH + kb];
        #pragma unroll
        for (int i = 0; i < 8; ++i) { b[i] = p[i]; b[i + 8] = p[i + 16]; }
        v8f acc = {};
        acc = __builtin_amdgcn_wmma_f32_16x16x32_f16(false, a, false, b,
                                                     (short)0, acc, false, false);
        const int   k  = t * 16 + col;
        const float cs = sCsq[k];
        #pragma unroll
        for (int j = 0; j < 8; ++j) {
            const float s = cs - 2.0f * acc[j];   // dist_sq minus the row-constant x_sq
            if (s < minv[j]) { minv[j] = s; mink[j] = k; }   // strict < keeps lowest k
        }
    }

    // cross-lane argmin within each 16-lane half (all k residues), tie -> lowest k
    #pragma unroll
    for (int off = 1; off < 16; off <<= 1) {
        #pragma unroll
        for (int j = 0; j < 8; ++j) {
            const float ov = __shfl_xor(minv[j], off, 32);
            const int   ok = __shfl_xor(mink[j], off, 32);
            if (ov < minv[j] || (ov == minv[j] && ok < mink[j])) { minv[j] = ov; mink[j] = ok; }
        }
    }
    // C/D layout: lanes 0-15 hold rows j, lanes 16-31 hold rows j+8
    if (col < 8) {
        const int rloc = w * 16 + ((lane < 16) ? col : col + 8);
        sLab[rloc] = mink[col];
    }
    __syncthreads();

    // gather centroids (fp32, L2-resident), write X_r & X_r_m & labels, error partials
    float err = 0.0f;
    #pragma unroll 1
    for (int r = 0; r < 16; ++r) {
        const int row = w * 16 + r;
        const int b   = brow0 + row;
        const int lab = sLab[row];
        const float cv = center[((size_t)m * K_CENT + lab) * D_SUB + lane];
        const size_t oidx = (size_t)b * D_TOT + m * D_SUB + lane;
        const float xv = X[oidx];
        out[OFF_XR + oidx]  = cv;
        out[OFF_XRM + oidx] = cv;
        const float dlt = cv - xv;
        err += dlt * dlt;
        if (lane == 0) out[OFF_LBL + (size_t)b * M_SUB + m] = (float)lab;
    }
    #pragma unroll
    for (int off = 16; off >= 1; off >>= 1) err += __shfl_xor(err, off, 32);
    if (lane == 0) sErr[w] = err;
    __syncthreads();
    if (tid == 0) {
        float s = 0.0f;
        #pragma unroll
        for (int i = 0; i < 8; ++i) s += sErr[i];
        partials[blockIdx.x] = s;
    }
}

// ---------------------------------------------------------------------------
// Kernel 4: deterministic fixed-order final reduction
__global__ __launch_bounds__(256) void finalize(const float* __restrict__ partials,
                                                float* __restrict__ out) {
    __shared__ double s[256];
    double acc = 0.0;
    for (int i = threadIdx.x; i < NBLK; i += 256) acc += (double)partials[i];
    s[threadIdx.x] = acc;
    __syncthreads();
    for (int off = 128; off >= 1; off >>= 1) {
        if (threadIdx.x < off) s[threadIdx.x] += s[threadIdx.x + off];
        __syncthreads();
    }
    // output = mean((X_r-X_p)^2) + mean((X_r_m-X)^2) = 2 * sum / (B*D)
    if (threadIdx.x == 0)
        out[OFF_OUT] = (float)(2.0 * s[0] / (double)((size_t)B_TOT * D_TOT));
}

// ---------------------------------------------------------------------------
extern "C" void kernel_launch(void* const* d_in, const int* in_sizes, int n_in,
                              void* d_out, int out_size, void* d_ws, size_t ws_size,
                              hipStream_t stream) {
    const float* X      = (const float*)d_in[0];
    const float* center = (const float*)d_in[1];
    float* out = (float*)d_out;

    _Float16* ch       = (_Float16*)d_ws;                          // 262144 B
    float*    csq      = (float*)((char*)d_ws + 262144);           //  16384 B
    float*    partials = (float*)((char*)d_ws + 262144 + 16384);   //   8192 B

    prep_center<<<512, 256, 0, stream>>>(center, ch, csq, out);
    copy_x<<<8192, 256, 0, stream>>>((const float4*)X, out);
    diffpq_compute<<<NBLK, THREADS, 0, stream>>>(X, center, ch, csq, out, partials);
    finalize<<<1, 256, 0, stream>>>(partials, out);
}
